// MultiHeadLocalAttention_48043504173474
// MI455X (gfx1250) — matrix operand order repaired and checked by reading, not compiled, not run on hardware
//
#include <hip/hip_runtime.h>

// ---------------------------------------------------------------------------
// Multi-head local (banded) attention for MI455X / gfx1250.
// Strategy: f16 WMMA (v_wmma_f32_16x16x32_f16) for all matmuls, f32 accumulate.
// The 268MB dense attn output write dominates (~11.5us @ 23.3TB/s) -> stream it
// with non-temporal b128 stores so L2 stays warm for GEMM operands.
// Softmax row reductions use DPP16 row_ror rotate-reduce (pure VALU, no LDS).
// ---------------------------------------------------------------------------

typedef _Float16 v16h __attribute__((ext_vector_type(16)));
typedef float    v8f  __attribute__((ext_vector_type(8)));
typedef float    v4f  __attribute__((ext_vector_type(4)));

#define T_SEQ 2048
#define NH    8
#define HS    64
#define CDIM  512
#define C3    1536
#define WIN   16

// K index of half-pair p (0..7) for the 16-bit A-matrix (16x32) lane layout:
// lanes 0-15: VGPR p holds K = 2p,2p+1; lanes 16-31: +16 (contiguous, matches B)
__device__ __forceinline__ int a16_k(int lane, int p) {
  int kb = (lane >> 4) << 4;
  return (2 * p) + kb;
}

__device__ __forceinline__ v8f wmma_f16(v16h a, v16h b, v8f c) {
  return __builtin_amdgcn_wmma_f32_16x16x32_f16(
      /*neg_a=*/false, a, /*neg_b=*/false, b,
      /*c_mod=*/(short)0, c, /*reuse_a=*/false, /*reuse_b=*/false);
}

// DPP16 row_ror:N — rotate within each 16-lane row (never crosses the half-wave
// boundary, matching the WMMA C-layout row domain). dpp_ctrl must be an ICE,
// hence the template parameter.
template <int N>
__device__ __forceinline__ float row_ror(float x) {
  int r = __builtin_amdgcn_update_dpp(0, __float_as_int(x),
                                      0x120 | N, 0xf, 0xf, true);
  return __int_as_float(r);
}

// Rotate-reduce over a 16-lane row: every lane ends with the row-wide result.
__device__ __forceinline__ float row_max16(float x) {
  x = fmaxf(x, row_ror<1>(x));
  x = fmaxf(x, row_ror<2>(x));
  x = fmaxf(x, row_ror<4>(x));
  x = fmaxf(x, row_ror<8>(x));
  return x;
}
__device__ __forceinline__ float row_sum16(float x) {
  x += row_ror<1>(x);
  x += row_ror<2>(x);
  x += row_ror<4>(x);
  x += row_ror<8>(x);
  return x;
}

// ---------------------------------------------------------------------------
// Helper kernels: f32->f16 convert, transpose-convert (for B operands), NT fill
// ---------------------------------------------------------------------------
__global__ void cvt_f32_to_f16(const float* __restrict__ in,
                               _Float16* __restrict__ out, int n) {
  int i = blockIdx.x * blockDim.x + threadIdx.x;
  if (i < n) out[i] = (_Float16)in[i];
}

// w: K x N row-major (f32) -> wt: N x K (f16). Makes WMMA B loads contiguous.
__global__ void transpose_cvt(const float* __restrict__ w,
                              _Float16* __restrict__ wt, int K, int N) {
  int id = blockIdx.x * blockDim.x + threadIdx.x;
  if (id >= K * N) return;
  int n = id / K;
  int k = id - n * K;
  wt[(size_t)n * K + k] = (_Float16)w[(size_t)k * N + n];
}

// Streaming zero-fill: 268MB > 192MB L2, so use non-temporal b128 stores.
__global__ void fill_zero_nt(v4f* __restrict__ p, long n4) {
  long i = blockIdx.x * (long)blockDim.x + threadIdx.x;
  long stride = (long)gridDim.x * blockDim.x;
  v4f z = {0.f, 0.f, 0.f, 0.f};
  for (; i < n4; i += stride) __builtin_nontemporal_store(z, p + i);
}

// ---------------------------------------------------------------------------
// WMMA GEMM: one wave computes a 64x64 tile of O = A(MxK) * B(KxN) + bias.
// A is MxK f16 row-major; BT is the transposed B, NxK f16 (contiguous K loads).
// 4x4 WMMA subtiles per K-step of 32.
// ---------------------------------------------------------------------------
template <bool OUT_F32>
__global__ void gemm_wmma(const _Float16* __restrict__ A,
                          const _Float16* __restrict__ BT,
                          const float* __restrict__ bias,
                          _Float16* __restrict__ O16,
                          float* __restrict__ O32,
                          int M, int N, int K) {
  const int lane  = threadIdx.x & 31;
  const int m0    = blockIdx.x * 64;
  const int n0    = blockIdx.y * 64;
  const int mrow  = lane & 15;
  const int khalf = lane >> 4;

  v8f acc[4][4] = {};

  for (int k0 = 0; k0 < K; k0 += 32) {
    v16h a[4];
#pragma unroll
    for (int ms = 0; ms < 4; ++ms) {
      const _Float16* ap = A + (size_t)(m0 + ms * 16 + mrow) * K + k0;
#pragma unroll
      for (int p = 0; p < 8; ++p) {
        int kk = a16_k(lane, p);
        a[ms][2 * p]     = ap[kk];
        a[ms][2 * p + 1] = ap[kk + 1];
      }
    }
#pragma unroll
    for (int ns = 0; ns < 4; ++ns) {
      int col = n0 + ns * 16 + mrow;  // B operand: N = lane&15
      const _Float16* bp = BT + (size_t)col * K + k0 + (khalf << 4);
      v16h b;
#pragma unroll
      for (int j = 0; j < 16; ++j) b[j] = bp[j];  // contiguous 32B -> b128 loads
#pragma unroll
      for (int ms = 0; ms < 4; ++ms)
        acc[ms][ns] = wmma_f16(a[ms], b, acc[ms][ns]);
    }
  }

  // Epilogue: C/D layout -> element [m][n]: m = r + 8*(lane>>4), n = lane&15
#pragma unroll
  for (int ms = 0; ms < 4; ++ms) {
#pragma unroll
    for (int ns = 0; ns < 4; ++ns) {
      int col = n0 + ns * 16 + (lane & 15);
      float bv = bias[col];
#pragma unroll
      for (int r = 0; r < 8; ++r) {
        int row = m0 + ms * 16 + r + (khalf << 3);
        float v = acc[ms][ns][r] + bv;
        if (OUT_F32) O32[(size_t)row * N + col] = v;
        else         O16[(size_t)row * N + col] = (_Float16)v;
      }
    }
  }
}

// ---------------------------------------------------------------------------
// Banded attention. One wave per (b, h, 16-query tile).
// Scores: Q(16x64) @ K^T(64x32) -> two 16x16 f32 tiles (4 WMMAs).
// Softmax per row over the 32-key stripe via DPP16 row_ror rotate-reduce.
// Probabilities -> dense attn output (stripe; fill kernel zeroed the rest),
// and -> LDS (f16) for C-layout -> A-layout re-layout.
// Out: P(16x32) @ V(32x64) -> 4 WMMAs, stored f16 for the projection GEMM.
// ---------------------------------------------------------------------------
__global__ void local_attn_kernel(const _Float16* __restrict__ qkv,  // (B*T) x 1536
                                  float* __restrict__ attn,          // B*H*T*T
                                  _Float16* __restrict__ o16) {      // (B*T) x 512
  const int lane = threadIdx.x & 31;
  int idx = blockIdx.x;
  int qt = idx & 127;          // T/16 = 128 tiles
  int h  = (idx >> 7) & 7;
  int b  = idx >> 10;
  const int hl   = lane >> 4;
  const int n    = lane & 15;
  const int kb16 = hl << 4;

  // ---- Q as A-operand (K = 64 -> two k-steps of 32) ----
  v16h aq0, aq1;
  {
    const _Float16* qp = qkv + (size_t)(b * T_SEQ + qt * 16 + n) * C3 + h * HS;
#pragma unroll
    for (int p = 0; p < 8; ++p) {
      int kk = a16_k(lane, p);
      aq0[2 * p]     = qp[kk];
      aq0[2 * p + 1] = qp[kk + 1];
      aq1[2 * p]     = qp[kk + 32];
      aq1[2 * p + 1] = qp[kk + 33];
    }
  }

  // ---- scores: two 16-key blocks, keys j in [qt*16-16, qt*16+15] ----
  v8f c[2] = {};
#pragma unroll
  for (int jb = 0; jb < 2; ++jb) {
    int key  = qt * 16 - 16 + jb * 16 + n;   // B operand: N = key column
    int keyc = key < 0 ? 0 : key;            // clamp; masked below
    const _Float16* kp = qkv + (size_t)(b * T_SEQ + keyc) * C3 + CDIM + h * HS;
    v16h bk0, bk1;
#pragma unroll
    for (int j = 0; j < 16; ++j) {
      bk0[j] = kp[kb16 + j];
      bk1[j] = kp[32 + kb16 + j];
    }
    c[jb] = wmma_f16(aq0, bk0, c[jb]);
    c[jb] = wmma_f16(aq1, bk1, c[jb]);
  }

  // ---- band mask + softmax (DPP rotate-reduce) + write probabilities ----
  __shared__ _Float16 P[16 * 32];
  const float scale = 0.125f;  // 1/sqrt(64)
#pragma unroll
  for (int r = 0; r < 8; ++r) {
    int m  = r + (hl << 3);
    int i  = qt * 16 + m;
    int j0 = qt * 16 - 16 + n;
    int j1 = qt * 16 + n;
    float s0 = (j0 >= 0 && j0 >= i - WIN && j0 <= i) ? c[0][r] * scale
                                                     : -__builtin_inff();
    float s1 = (j1 >= i - WIN && j1 <= i) ? c[1][r] * scale
                                          : -__builtin_inff();
    float mx = row_max16(fmaxf(s0, s1));
    float e0 = __expf(s0 - mx);
    float e1 = __expf(s1 - mx);
    float sm = row_sum16(e0 + e1);
    float inv = 1.0f / sm;
    float p0 = e0 * inv;
    float p1 = e1 * inv;

    size_t abase = ((size_t)((b * NH + h) * T_SEQ + i)) * T_SEQ;
    if (j0 >= 0) attn[abase + j0] = p0;
    attn[abase + j1] = p1;

    P[m * 32 + n]      = (_Float16)p0;
    P[m * 32 + 16 + n] = (_Float16)p1;
  }
  __syncthreads();

  // ---- out = P(16x32) @ V(32x64), one k-step of 32 ----
  v16h pa;
#pragma unroll
  for (int p = 0; p < 8; ++p) {
    int kk = a16_k(lane, p);
    pa[2 * p]     = P[n * 32 + kk];
    pa[2 * p + 1] = P[n * 32 + kk + 1];
  }
  v8f oc[4] = {};
#pragma unroll
  for (int ns = 0; ns < 4; ++ns) {
    v16h bvv;
#pragma unroll
    for (int j = 0; j < 16; ++j) {
      int key  = qt * 16 - 16 + kb16 + j;
      int keyc = key < 0 ? 0 : key;  // weight is 0 there
      bvv[j] = qkv[(size_t)(b * T_SEQ + keyc) * C3 + 2 * CDIM + h * HS +
                   ns * 16 + n];
    }
    oc[ns] = wmma_f16(pa, bvv, oc[ns]);
  }
#pragma unroll
  for (int ns = 0; ns < 4; ++ns) {
#pragma unroll
    for (int r = 0; r < 8; ++r) {
      int i = qt * 16 + r + (hl << 3);
      int d = ns * 16 + n;
      o16[(size_t)(b * T_SEQ + i) * CDIM + h * HS + d] = (_Float16)oc[ns][r];
    }
  }
}

// ---------------------------------------------------------------------------
extern "C" void kernel_launch(void* const* d_in, const int* in_sizes, int n_in,
                              void* d_out, int out_size, void* d_ws,
                              size_t ws_size, hipStream_t stream) {
  (void)in_sizes; (void)n_in; (void)out_size; (void)ws_size;

  const float* x      = (const float*)d_in[0];
  const float* w_attn = (const float*)d_in[1];
  const float* b_attn = (const float*)d_in[2];
  const float* w_proj = (const float*)d_in[3];
  const float* b_proj = (const float*)d_in[4];

  const int B = 2, T = T_SEQ, C = CDIM, C3v = C3, H = NH;
  const int M = B * T;  // 4096 token rows

  // workspace layout (f16), ~22 MiB total
  _Float16* ws    = (_Float16*)d_ws;
  _Float16* x16   = ws;                              // M*C
  _Float16* waT16 = x16 + (size_t)M * C;             // C3*C (w_attn transposed)
  _Float16* wpT16 = waT16 + (size_t)C3v * C;         // C*C  (w_proj transposed)
  _Float16* qkv16 = wpT16 + (size_t)C * C;           // M*C3
  _Float16* o16   = qkv16 + (size_t)M * C3v;         // M*C

  float* out  = (float*)d_out;                       // (B,T,C) f32
  float* attn = out + (size_t)M * C;                 // (B,H,T,T) f32

  // 1) precision conversion (f32 -> f16), weights transposed for B-operand
  cvt_f32_to_f16<<<(M * C + 255) / 256, 256, 0, stream>>>(x, x16, M * C);
  transpose_cvt<<<(C * C3v + 255) / 256, 256, 0, stream>>>(w_attn, waT16, C, C3v);
  transpose_cvt<<<(C * C + 255) / 256, 256, 0, stream>>>(w_proj, wpT16, C, C);

  // 2) zero the dense attn tensor (268 MB; bandwidth-dominant; NT streaming)
  long n4 = ((long)B * H * T * T) / 4;
  fill_zero_nt<<<4096, 256, 0, stream>>>((v4f*)attn, n4);

  // 3) QKV GEMM: (M x C) @ (C x 3C) + b_attn -> qkv16 (f16)
  dim3 g1(M / 64, C3v / 64);
  gemm_wmma<false><<<g1, 32, 0, stream>>>(x16, waT16, b_attn, qkv16, nullptr,
                                          M, C3v, C);

  // 4) banded attention (writes attn stripe + o16)
  local_attn_kernel<<<B * H * (T / 16), 32, 0, stream>>>(qkv16, attn, o16);

  // 5) projection GEMM: (M x C) @ (C x C) + b_proj -> out (f32)
  dim3 g2(M / 64, C / 64);
  gemm_wmma<true><<<g2, 32, 0, stream>>>(o16, wpT16, b_proj, nullptr, out,
                                         M, C, C);
}
